// MultiHeadedTrilinearAttention_70411693851082
// MI455X (gfx1250) — compile-verified, matrix operand order checked by reading, not run on hardware
//
#include <hip/hip_runtime.h>
#include <hip/hip_bf16.h>

typedef __attribute__((ext_vector_type(16))) _Float16 v16h;
typedef __attribute__((ext_vector_type(8)))  _Float16 v8h;
typedef __attribute__((ext_vector_type(8)))  float    v8f;
typedef __attribute__((ext_vector_type(4)))  unsigned int u32x4;
typedef __attribute__((ext_vector_type(8)))  unsigned int u32x8;

#define DIMD 768
#define HEADS 12
#define HW 64
#define SEQ 128

union Frag16 { v16h v; v8h h[2]; };

// Load a 16x32 f16 fragment from a row-major matrix (leading dim ld) at (row0, k0).
// CDNA5 16-bit A/B fragment layout: lane L holds row (L&15); with hh=L>>4,
// vector elems 0..7 = K k0+8*hh .. +7, elems 8..15 = K k0+16+8*hh .. +7.
__device__ __forceinline__ v16h load_frag_rm(const _Float16* __restrict__ base, int ld,
                                             int row0, int k0, int lane) {
  const int m  = lane & 15;
  const int hh = lane >> 4;
  const _Float16* p = base + (size_t)(row0 + m) * ld + (k0 + 8 * hh);
  Frag16 f;
  f.h[0] = *(const v8h*)(p);
  f.h[1] = *(const v8h*)(p + 16);
  return f.v;
}

// Same, from an LDS-resident tile with fixed 64-half row stride.
__device__ __forceinline__ v16h load_frag_lds(const _Float16* base, int row0, int k0, int lane) {
  const int m  = lane & 15;
  const int hh = lane >> 4;
  const _Float16* p = base + (row0 + m) * HW + (k0 + 8 * hh);
  Frag16 f;
  f.h[0] = *(const v8h*)(p);
  f.h[1] = *(const v8h*)(p + 16);
  return f.v;
}

// Same as load_frag_rm, rows elementwise-multiplied by a 64-wide f16 vector in LDS.
__device__ __forceinline__ v16h load_frag_scaled(const _Float16* __restrict__ base, int ld,
                                                 int row0, int k0, int lane,
                                                 const _Float16* vrow) {
  const int m  = lane & 15;
  const int hh = lane >> 4;
  const _Float16* p = base + (size_t)(row0 + m) * ld + (k0 + 8 * hh);
  Frag16 f;
  f.h[0] = (*(const v8h*)(p))      * (*(const v8h*)(vrow + k0 + 8 * hh));
  f.h[1] = (*(const v8h*)(p + 16)) * (*(const v8h*)(vrow + k0 + 16 + 8 * hh));
  return f.v;
}

// Tensor Data Mover: DMA a 2D tile (tile_d0 x tile_d1 elements, 2-byte elements,
// row stride stride_d0 elements) from global memory into LDS at lds_byte_addr.
// D# bitfield packing per CDNA5 ISA ch.8 (groups 0/1; groups 2/3 zero for 2D).
__device__ __forceinline__ void tdm_load_tile_2d_f16(unsigned lds_byte_addr,
                                                     const void* gptr,
                                                     unsigned tensor_d0, unsigned tensor_d1,
                                                     unsigned tile_d0,   unsigned tile_d1,
                                                     unsigned stride_d0) {
  const unsigned long long ga = (unsigned long long)(size_t)gptr;
  u32x4 g0;
  g0[0] = 1u;                                                 // count=1 (valid), user mode
  g0[1] = lds_byte_addr;                                      // lds_addr [63:32]
  g0[2] = (unsigned)(ga & 0xffffffffu);                       // global_addr [95:64]
  g0[3] = (unsigned)((ga >> 32) & 0x01ffffffu) | (2u << 30);  // global_addr hi + type=2
  u32x8 g1;
  g1[0] = (1u << 16);                                         // data_size=1 -> 2 bytes
  g1[1] = (tensor_d0 & 0xffffu) << 16;                        // tensor_dim0 [79:48] lo
  g1[2] = (tensor_d0 >> 16) | ((tensor_d1 & 0xffffu) << 16);  // tensor_dim0 hi | tensor_dim1 lo
  g1[3] = (tensor_d1 >> 16) | (tile_d0 << 16);                // tensor_dim1 hi | tile_dim0
  g1[4] = tile_d1;                                            // tile_dim1; tile_dim2=0
  g1[5] = stride_d0;                                          // tensor_dim0_stride lo32
  g1[6] = 0u;                                                 // stride hi | dim1_stride lo
  g1[7] = 0u;
  u32x4 gz = {0u, 0u, 0u, 0u};                                // groups 2/3: unused dims = 0
  asm volatile("tensor_load_to_lds %0, %1, %2, %3"
               :: "s"(g0), "s"(g1), "s"(gz), "s"(gz)
               : "memory");
}

__global__ void cvt_kernel(const float* __restrict__ in, _Float16* __restrict__ out, int n) {
  int i = blockIdx.x * blockDim.x + threadIdx.x;
  if (i < n) out[i] = (_Float16)in[i];
}

// Wt[n][k] = (f16) W[k][n]
__global__ void transpose_cvt_kernel(const float* __restrict__ W, _Float16* __restrict__ Wt, int D) {
  __shared__ float tile[32][33];
  const int bx = blockIdx.x * 32;
  const int by = blockIdx.y * 32;
  const int tx = threadIdx.x;
  const int ty = threadIdx.y;
  #pragma unroll
  for (int i = ty; i < 32; i += 8) tile[i][tx] = W[(size_t)(by + i) * D + bx + tx];
  __syncthreads();
  #pragma unroll
  for (int i = ty; i < 32; i += 8) Wt[(size_t)(bx + i) * D + by + tx] = (_Float16)tile[tx][i];
}

// C[M,N] = A[M,K] (f16, rm) @ Bt[N,K]^T (f16, rm) + bias. 4 waves/block; each wave
// does one 16-row M tile x 64 columns. B fragments preloaded so the 8 global_load_b128
// per k-step clause together and overlap the previous WMMA chain.
template <bool F16OUT>
__global__ void __launch_bounds__(128) gemm_wmma_kernel(
    const _Float16* __restrict__ A, const _Float16* __restrict__ Bt,
    const float* __restrict__ bias, void* __restrict__ out,
    int M, int N, int K)
{
  const int lane = threadIdx.x & 31;
  const int wave = threadIdx.x >> 5;
  const int m0 = blockIdx.y * 16;
  const int n0 = blockIdx.x * 256 + wave * 64;

  v8f acc[4];
  #pragma unroll
  for (int j = 0; j < 4; ++j)
    #pragma unroll
    for (int r = 0; r < 8; ++r) acc[j][r] = 0.0f;

  for (int k0 = 0; k0 < K; k0 += 32) {
    v16h afr = load_frag_rm(A, K, m0, k0, lane);
    v16h bfr[4];
    #pragma unroll
    for (int j = 0; j < 4; ++j) bfr[j] = load_frag_rm(Bt, K, n0 + 16 * j, k0, lane);
    #pragma unroll
    for (int j = 0; j < 4; ++j)
      acc[j] = __builtin_amdgcn_wmma_f32_16x16x32_f16(
          false, afr, false, bfr[j], (short)0, acc[j], false, false);
  }

  const int mb = m0 + 8 * (lane >> 4);
  #pragma unroll
  for (int j = 0; j < 4; ++j) {
    const int n = n0 + 16 * j + (lane & 15);
    const float bv = bias[n];
    #pragma unroll
    for (int r = 0; r < 8; ++r) {
      const float val = acc[j][r] + bv;
      if (F16OUT) ((_Float16*)out)[(size_t)(mb + r) * N + n] = (_Float16)val;
      else        ((float*)out)[(size_t)(mb + r) * N + n]    = val;
    }
  }
}

// One workgroup per 4096-score softmax chunk g in [0,6144):
//   h = g/512, v = (g>>2)&127, q-block = (g&3)*32;  b2 = g/12, hh = g%12.
// scores[q',a] = (1/8) * (Qh-rows ∘ vh-row) @ Ah^T  -> 32x128 tile via WMMA.
// The 128x64 Ah head-tile is DMA'd into LDS once per workgroup by the TDM.
__global__ void __launch_bounds__(128) trilinear_fused_kernel(
    const _Float16* __restrict__ vp16, const _Float16* __restrict__ qp16,
    const _Float16* __restrict__ ap16,
    const float* __restrict__ vo, const float* __restrict__ qo,
    const float* __restrict__ ao, float* __restrict__ out)
{
  const int g  = blockIdx.x;
  const int h  = g >> 9;
  const int v  = (g >> 2) & 127;
  const int q0 = (g & 3) * 32;
  const int tid  = threadIdx.x;
  const int lane = tid & 31;
  const int wave = tid >> 5;

  __shared__ __align__(16) _Float16 atile[SEQ * HW];   // 16 KB: Ah tile via TDM
  __shared__ __align__(16) _Float16 vrow[64];
  __shared__ float sc[4096];
  __shared__ float red[128];
  __shared__ float Ssum[48];

  // Kick the TDM from one wave: Ah tile = ap16[a*768 + h*64 + w], a<128, w<64.
  if (wave == 0) {
    tdm_load_tile_2d_f16((unsigned)(size_t)&atile[0],
                         ap16 + (size_t)h * HW,
                         /*tensor_d0=*/HW, /*tensor_d1=*/SEQ,
                         /*tile_d0=*/HW,   /*tile_d1=*/SEQ,
                         /*stride_d0=*/DIMD);
    __builtin_amdgcn_s_wait_tensorcnt(0);
  }
  if (tid < 64) vrow[tid] = vp16[(size_t)v * DIMD + h * HW + tid];
  if (tid < 48) Ssum[tid] = 0.0f;
  __syncthreads();

  const _Float16* qbase = qp16 + (size_t)q0 * DIMD + h * HW;

  const int mi  = wave >> 1;       // which 16-row half of the 32 q'
  const int ni0 = (wave & 1) * 4;  // which 4 of 8 a-column tiles

  v8f acc[4];
  #pragma unroll
  for (int j = 0; j < 4; ++j)
    #pragma unroll
    for (int r = 0; r < 8; ++r) acc[j][r] = 0.0f;

  #pragma unroll
  for (int kk = 0; kk < 2; ++kk) {
    v16h afr = load_frag_scaled(qbase, DIMD, mi * 16, kk * 32, lane, vrow);
    v16h bfr[4];
    #pragma unroll
    for (int j = 0; j < 4; ++j) bfr[j] = load_frag_lds(atile, (ni0 + j) * 16, kk * 32, lane);
    #pragma unroll
    for (int j = 0; j < 4; ++j)
      acc[j] = __builtin_amdgcn_wmma_f32_16x16x32_f16(
          false, afr, false, bfr[j], (short)0, acc[j], false, false);
  }

  const int mbase = mi * 16 + 8 * (lane >> 4);
  #pragma unroll
  for (int j = 0; j < 4; ++j) {
    const int n = (ni0 + j) * 16 + (lane & 15);
    #pragma unroll
    for (int r = 0; r < 8; ++r) sc[(mbase + r) * 128 + n] = acc[j][r] * 0.125f;
  }
  __syncthreads();

  // ---- softmax over all 4096 ----
  float mx = -3.0e38f;
  for (int i = tid; i < 4096; i += 128) mx = fmaxf(mx, sc[i]);
  red[tid] = mx;
  __syncthreads();
  #pragma unroll
  for (int s = 64; s > 0; s >>= 1) {
    if (tid < s) red[tid] = fmaxf(red[tid], red[tid + s]);
    __syncthreads();
  }
  const float gmax = red[0];
  __syncthreads();

  float lsum = 0.0f;
  for (int i = tid; i < 4096; i += 128) {
    const float e = __expf(sc[i] - gmax);
    sc[i] = e;
    lsum += e;
  }
  red[tid] = lsum;
  __syncthreads();
  #pragma unroll
  for (int s = 64; s > 0; s >>= 1) {
    if (tid < s) red[tid] += red[tid + s];
    __syncthreads();
  }
  const float invTotal = 1.0f / red[0];

  // ---- the three 16-sum families over r = vv*256 + qq*16 + aa ----
  {
    const int sub = tid >> 3;  // 0..15
    const int off = tid & 7;   // 0..7
    float s0 = 0.f, s1 = 0.f, s2 = 0.f;
    for (int u = off; u < 256; u += 8) s0 += sc[sub * 256 + u];                    // Sv
    #pragma unroll
    for (int vvi = 0; vvi < 16; ++vvi) {
      s1 += sc[vvi * 256 + sub * 16 + off] + sc[vvi * 256 + sub * 16 + off + 8];   // Sq
      s2 += sc[vvi * 256 + off * 16 + sub] + sc[vvi * 256 + (off + 8) * 16 + sub]; // Sa
    }
    atomicAdd(&Ssum[sub],      s0);
    atomicAdd(&Ssum[16 + sub], s1);
    atomicAdd(&Ssum[32 + sub], s2);
  }
  __syncthreads();

  // ---- 48 owned output elements: out[which][b2*192 + idx*12 + hh] ----
  if (tid < 48) {
    const int which = tid >> 4;
    const int idx   = tid & 15;
    const int b2 = g / 12;
    const int hh = g % 12;
    const int f  = 16 * g + idx;          // flat index into heads(vo/qo/ao)
    const int hp = f >> 13;               // head
    const int sp = (f >> 6) & 127;        // token
    const int wp = f & 63;                // within-head col
    const float* proj = (which == 0) ? vo : ((which == 1) ? qo : ao);
    const float pv = proj[(size_t)sp * DIMD + hp * HW + wp];
    out[which * (SEQ * DIMD) + b2 * 192 + idx * 12 + hh] = pv * Ssum[tid] * invTotal;
  }
}

extern "C" void kernel_launch(void* const* d_in, const int* in_sizes, int n_in,
                              void* d_out, int out_size, void* d_ws, size_t ws_size,
                              hipStream_t stream) {
  (void)in_sizes; (void)n_in; (void)out_size; (void)ws_size;
  const float* v   = (const float*)d_in[0];
  const float* q   = (const float*)d_in[1];
  const float* a   = (const float*)d_in[2];
  // d_in[3..5] = masks (unused by the reference forward)
  const float* Wsrc[6] = { (const float*)d_in[6],  (const float*)d_in[8],
                           (const float*)d_in[10], (const float*)d_in[12],
                           (const float*)d_in[14], (const float*)d_in[16] };
  const float* bv  = (const float*)d_in[7];
  const float* bq  = (const float*)d_in[9];
  const float* ba  = (const float*)d_in[11];
  const float* bvo = (const float*)d_in[13];
  const float* bqo = (const float*)d_in[15];
  const float* bao = (const float*)d_in[17];

  char* ws = (char*)d_ws;
  size_t off = 0;
  auto alloc = [&](size_t bytes) {
    char* p = ws + off;
    off += (bytes + 255) & ~(size_t)255;
    return p;
  };
  const size_t WBYTES = (size_t)DIMD * DIMD * sizeof(_Float16);
  const size_t XH     = (size_t)SEQ * DIMD * sizeof(_Float16);
  const size_t XF     = (size_t)SEQ * DIMD * sizeof(float);
  _Float16* Wt[6];
  for (int i = 0; i < 6; ++i) Wt[i] = (_Float16*)alloc(WBYTES);
  _Float16* v16  = (_Float16*)alloc(XH);
  _Float16* q16  = (_Float16*)alloc(XH);
  _Float16* a16  = (_Float16*)alloc(XH);
  _Float16* vp16 = (_Float16*)alloc(XH);
  _Float16* qp16 = (_Float16*)alloc(XH);
  _Float16* ap16 = (_Float16*)alloc(XH);
  float* vo = (float*)alloc(XF);
  float* qo = (float*)alloc(XF);
  float* ao = (float*)alloc(XF);

  {
    dim3 b(256), g((SEQ * DIMD + 255) / 256);
    cvt_kernel<<<g, b, 0, stream>>>(v, v16, SEQ * DIMD);
    cvt_kernel<<<g, b, 0, stream>>>(q, q16, SEQ * DIMD);
    cvt_kernel<<<g, b, 0, stream>>>(a, a16, SEQ * DIMD);
  }
  {
    dim3 b(32, 8), g(DIMD / 32, DIMD / 32);
    for (int i = 0; i < 6; ++i)
      transpose_cvt_kernel<<<g, b, 0, stream>>>(Wsrc[i], Wt[i], DIMD);
  }
  {
    dim3 b(128), g(DIMD / 256, SEQ / 16);  // (3, 8)
    gemm_wmma_kernel<true ><<<g, b, 0, stream>>>(v16,  Wt[0], bv,  vp16, SEQ, DIMD, DIMD);
    gemm_wmma_kernel<true ><<<g, b, 0, stream>>>(q16,  Wt[1], bq,  qp16, SEQ, DIMD, DIMD);
    gemm_wmma_kernel<true ><<<g, b, 0, stream>>>(a16,  Wt[2], ba,  ap16, SEQ, DIMD, DIMD);
    gemm_wmma_kernel<false><<<g, b, 0, stream>>>(vp16, Wt[3], bvo, vo,   SEQ, DIMD, DIMD);
    gemm_wmma_kernel<false><<<g, b, 0, stream>>>(qp16, Wt[4], bqo, qo,   SEQ, DIMD, DIMD);
    gemm_wmma_kernel<false><<<g, b, 0, stream>>>(ap16, Wt[5], bao, ao,   SEQ, DIMD, DIMD);
  }
  trilinear_fused_kernel<<<dim3(6144), dim3(128), 0, stream>>>(
      vp16, qp16, ap16, vo, qo, ao, (float*)d_out);
}
